// CausalSelfAttention_83270825935204
// MI455X (gfx1250) — matrix operand order repaired and checked by reading, not compiled, not run on hardware
//
#include <hip/hip_runtime.h>

// ---------------------------------------------------------------------------
// CDNA5 (gfx1250) causal self-attention, bf16 WMMA pipeline, fp32 accumulate.
// All GEMM-shaped work goes through v_wmma_f32_16x16x32_bf16 (wave32).
// Staging uses GLOBAL_LOAD_ASYNC_TO_LDS_B128 (ASYNCcnt) with double-buffered
// LDS when the toolchain exposes the builtin; otherwise a register-prefetch
// software pipeline with the same overlap structure.
// ---------------------------------------------------------------------------

typedef __bf16 bf16_t;
typedef __attribute__((ext_vector_type(16))) __bf16 v16bf;
typedef __attribute__((ext_vector_type(8)))  float  v8f;
typedef __attribute__((ext_vector_type(4)))  __bf16 v4bf;
typedef __attribute__((ext_vector_type(4)))  float  v4f;

#if defined(__has_builtin)
#if __has_builtin(__builtin_amdgcn_global_load_async_to_lds_b128) && \
    __has_builtin(__builtin_amdgcn_s_wait_asynccnt)
#define HAVE_ASYNC_LDS 1
#endif
#endif
#ifndef HAVE_ASYNC_LDS
#define HAVE_ASYNC_LDS 0
#endif

#if HAVE_ASYNC_LDS
// Builtin signature (from hipcc diagnostic): param0 is a pointer to a 4-int
// vector in the global ("__device__", address_space(1)) AS; param1 analogous
// in LDS (address_space(3)); then imm offset + cpol.
typedef int v4i __attribute__((__vector_size__(4 * sizeof(int))));
typedef __attribute__((address_space(1))) v4i gas_v4i;
typedef __attribute__((address_space(3))) v4i las_v4i;
__device__ __forceinline__ void async_ld128(const void* g, void* l) {
    __builtin_amdgcn_global_load_async_to_lds_b128(
        (gas_v4i*)g, (las_v4i*)l, /*offset=*/0, /*cpol=*/0);
}
#endif

union Frag {
    v16bf v;
    uint4 q[2];
};

__device__ __forceinline__ v8f wmma_bf16(const Frag& a, const Frag& b, v8f c) {
    return __builtin_amdgcn_wmma_f32_16x16x32_bf16(
        /*neg_a=*/false, a.v, /*neg_b=*/false, b.v,
        /*c_mod=*/(short)0, c, /*reuse_a=*/false, /*reuse_b=*/false);
}

// ---------------------------------------------------------------------------
// fp32 -> bf16 elementwise convert (vectorized x4)
// ---------------------------------------------------------------------------
__global__ __launch_bounds__(256) void k_f32_to_bf16(const float* __restrict__ src,
                                                     bf16_t* __restrict__ dst, int n) {
    int i = (blockIdx.x * 256 + threadIdx.x) * 4;
    if (i >= n) return;
    v4f f = *(const v4f*)(src + i);
    v4bf o;
    o[0] = (bf16_t)f[0]; o[1] = (bf16_t)f[1];
    o[2] = (bf16_t)f[2]; o[3] = (bf16_t)f[3];
    *(v4bf*)(dst + i) = o;
}

// ---------------------------------------------------------------------------
// Weight transpose + convert: W[k][n] (K=1024 rows) -> WT[n][k] bf16.
// [N,K] layout makes WMMA B-fragment loads contiguous (two runs of 8 consecutive K).
// ---------------------------------------------------------------------------
__global__ __launch_bounds__(256) void k_w_transpose_bf16(const float* __restrict__ W,
                                                          bf16_t* __restrict__ WT, int N) {
    int i = blockIdx.x * 256 + threadIdx.x;     // over N*1024
    if (i >= N * 1024) return;
    int n = i >> 10;
    int k = i & 1023;
    WT[i] = (bf16_t)W[(size_t)k * N + n];
}

// ---------------------------------------------------------------------------
// bf16 GEMM: Out[M,N] = A[M,K] * BT[N,K]^T, fp32 accumulate.
// Block tile 128x128, 8 waves (4x2), wave tile 32x64 = 8 WMMA tiles, BK=32.
// Double-buffered async LDS staging (or register-prefetch fallback).
// ---------------------------------------------------------------------------
template <bool OUT_BF16>
__global__ __launch_bounds__(256) void k_gemm_bf16(const bf16_t* __restrict__ A,
                                                   const bf16_t* __restrict__ BT,
                                                   void* __restrict__ Out,
                                                   int M, int N, int K) {
#if HAVE_ASYNC_LDS
    __shared__ uint4 smem[2048];                 // 2 x (A 512 | B 512) uint4
#else
    __shared__ uint4 smem[1024];                 // A 512 | B 512
#endif
    const int tid  = threadIdx.x;
    const int lane = tid & 31;
    const int l15  = lane & 15;
    const int hl   = (lane < 16) ? 0 : 1;
    const int wave = tid >> 5;
    const int wm   = (wave >> 1) * 32;           // wave M offset within block tile
    const int wn   = (wave & 1) * 64;            // wave N offset within block tile
    const int bm   = blockIdx.y * 128;
    const int bn   = blockIdx.x * 128;

    const int r0 = tid >> 2;                     // staging row (0..63), +64 for s=1
    const int c0 = (tid & 3) * 8;                // staging col element offset

#if HAVE_ASYNC_LDS
    auto issue = [&](int buf, int k0) {
        uint4* base = smem + buf * 1024;
#pragma unroll
        for (int s = 0; s < 2; s++) {
            int idx = tid + s * 256;
            int row = r0 + s * 64;
            async_ld128(A  + (size_t)(bm + row) * K + k0 + c0, base + idx);
            async_ld128(BT + (size_t)(bn + row) * K + k0 + c0, base + 512 + idx);
        }
    };
    issue(0, 0);
    int buf = 0;
#else
    uint4 ra[2], rb[2];
    auto gload = [&](int k0) {
#pragma unroll
        for (int s = 0; s < 2; s++) {
            int row = r0 + s * 64;
            ra[s] = *(const uint4*)(A  + (size_t)(bm + row) * K + k0 + c0);
            rb[s] = *(const uint4*)(BT + (size_t)(bn + row) * K + k0 + c0);
        }
    };
    gload(0);
#endif

    v8f acc[2][4] = {};

    for (int k0 = 0; k0 < K; k0 += 32) {
#if HAVE_ASYNC_LDS
        __builtin_amdgcn_s_wait_asynccnt(0);     // own panel landed in LDS
        __syncthreads();                         // everyone's panel landed
        const bf16_t* As = (const bf16_t*)(smem + buf * 1024);
        const bf16_t* Bs = (const bf16_t*)(smem + buf * 1024 + 512);
        if (k0 + 32 < K) issue(buf ^ 1, k0 + 32);  // overlap next panel with WMMA
        buf ^= 1;
#else
        __syncthreads();                         // previous compute done reading LDS
#pragma unroll
        for (int s = 0; s < 2; s++) {
            smem[tid + s * 256]       = ra[s];
            smem[tid + s * 256 + 512] = rb[s];
        }
        __syncthreads();
        const bf16_t* As = (const bf16_t*)smem;
        const bf16_t* Bs = (const bf16_t*)(smem + 512);
        if (k0 + 32 < K) gload(k0 + 32);         // overlap next panel with WMMA
        if (k0 + 64 < K) {
            __builtin_prefetch(A  + (size_t)(bm + r0) * K + k0 + 64 + c0, 0, 3);
            __builtin_prefetch(BT + (size_t)(bn + r0) * K + k0 + 64 + c0, 0, 3);
        }
#endif

        Frag af[2], bfr[4];
#pragma unroll
        for (int mt = 0; mt < 2; mt++) {
            int r = wm + mt * 16 + l15;          // A fragment: lane = M row
            int base = hl * 8;                   // K in {0-7,16-23} / {8-15,24-31}
            af[mt].q[0] = *(const uint4*)(As + r * 32 + base);
            af[mt].q[1] = *(const uint4*)(As + r * 32 + base + 16);
        }
#pragma unroll
        for (int nt = 0; nt < 4; nt++) {
            int r = wn + nt * 16 + l15;          // B fragment: lane = N col
            int base = hl * 8;                   // K in {0-7,16-23} / {8-15,24-31}
            bfr[nt].q[0] = *(const uint4*)(Bs + r * 32 + base);
            bfr[nt].q[1] = *(const uint4*)(Bs + r * 32 + base + 16);
        }
#pragma unroll
        for (int mt = 0; mt < 2; mt++)
#pragma unroll
            for (int nt = 0; nt < 4; nt++)
                acc[mt][nt] = wmma_bf16(af[mt], bfr[nt], acc[mt][nt]);
    }

#pragma unroll
    for (int mt = 0; mt < 2; mt++) {
#pragma unroll
        for (int nt = 0; nt < 4; nt++) {
            int n = bn + wn + nt * 16 + l15;
#pragma unroll
            for (int r = 0; r < 8; r++) {
                int m = bm + wm + mt * 16 + r + hl * 8;
                float vv = acc[mt][nt][r];
                if (OUT_BF16) ((bf16_t*)Out)[(size_t)m * N + n] = (bf16_t)vv;
                else          ((float*)Out)[(size_t)m * N + n]  = vv;
            }
        }
    }
}

// ---------------------------------------------------------------------------
// RoPE + scatter: qkv[B*T,3072] bf16 -> Q[b,h,t,d], K[b,h,t,d], V^T[b,h,d,t].
// ---------------------------------------------------------------------------
__global__ __launch_bounds__(256) void k_rope_scatter(const bf16_t* __restrict__ qkv,
                                                      bf16_t* __restrict__ Qr,
                                                      bf16_t* __restrict__ Kr,
                                                      bf16_t* __restrict__ Vt) {
    int idx = blockIdx.x * 256 + threadIdx.x;    // ((b*2048+t)*16+h)*32+i, 2^21 total
    int i  = idx & 31;
    int h  = (idx >> 5) & 15;
    int t  = (idx >> 9) & 2047;
    int b  = idx >> 20;
    int d0 = i * 2;

    size_t row = (size_t)(b * 2048 + t) * 3072;
    int col = h * 64 + d0;
    float q0 = (float)qkv[row + col],        q1 = (float)qkv[row + col + 1];
    float k0 = (float)qkv[row + 1024 + col], k1 = (float)qkv[row + 1024 + col + 1];
    bf16_t v0 = qkv[row + 2048 + col],       v1 = qkv[row + 2048 + col + 1];

    float ang = (float)t * __powf(10000.0f, -((float)d0) * (1.0f / 64.0f));
    float s, c;
    __sincosf(ang, &s, &c);

    size_t obase = ((size_t)(b * 16 + h) * 2048 + t) * 64 + d0;
    Qr[obase]     = (bf16_t)(q0 * c - q1 * s);
    Qr[obase + 1] = (bf16_t)(q1 * c + q0 * s);
    Kr[obase]     = (bf16_t)(k0 * c - k1 * s);
    Kr[obase + 1] = (bf16_t)(k1 * c + k0 * s);

    size_t vbase = ((size_t)(b * 16 + h) * 64 + d0) * 2048 + t;
    Vt[vbase]        = v0;
    Vt[vbase + 2048] = v1;
}

// ---------------------------------------------------------------------------
// Flash attention, S^T formulation. Grid: (T/128, B*H). 8 waves/block, each
// wave owns 16 q-rows. Per 32-kt step: S^T = K*Q^T (4 WMMAs) -> per-lane
// online softmax (one shfl_xor(16) per reduction) -> O^T += V^T*P^T (4 WMMAs),
// P^T rebuilt in-register with cross-half shuffles. All LDS fragment loads
// for a WMMA burst are issued before the burst to amortize DScnt waits.
// ---------------------------------------------------------------------------
__global__ __launch_bounds__(256) void k_attn(const bf16_t* __restrict__ Qr,
                                              const bf16_t* __restrict__ Kr,
                                              const bf16_t* __restrict__ Vt,
                                              bf16_t* __restrict__ Y) {
#if HAVE_ASYNC_LDS
    __shared__ uint4 smem[1024];                 // 2 x (K 32x64 | V^T 64x32)
#else
    __shared__ uint4 smem[512];                  // K 32x64 | V^T 64x32
#endif
    const int tid = threadIdx.x, lane = tid & 31, wave = tid >> 5;
    const int l15 = lane & 15, hl = (lane < 16) ? 0 : 1;
    const int bh = blockIdx.y;                   // b*16+h
    const int b = bh >> 4, h = bh & 15;
    const int q0blk = blockIdx.x * 128;
    const int qw = q0blk + wave * 16;

    const bf16_t* Kb = Kr + (size_t)bh * 2048 * 64;
    const bf16_t* Vb = Vt + (size_t)bh * 64 * 2048;
    const bf16_t* Qb = Qr + (size_t)bh * 2048 * 64;

    const int qlane = qw + l15;                  // this lane's q row (B-operand N index)

    const int krow = tid >> 3, kc = (tid & 7) * 8;   // K staging: 32 rows x 8 x 16B
    const int vrow = tid >> 2, vc = (tid & 3) * 8;   // V staging: 64 rows x 4 x 16B

    // Q as B operand of S^T: lane = q col, elements = d in {0-7,16-23} / {8-15,24-31}
    Frag qf[2];
#pragma unroll
    for (int s = 0; s < 2; s++) {
        int dbase = s * 32 + hl * 8;
        qf[s].q[0] = *(const uint4*)(Qb + (size_t)qlane * 64 + dbase);
        qf[s].q[1] = *(const uint4*)(Qb + (size_t)qlane * 64 + dbase + 16);
    }

    const int niter = (q0blk >> 5) + 4;          // kt covers [0, q0blk+128)

#if HAVE_ASYNC_LDS
    auto issue_kv = [&](int buf, int kt0) {
        uint4* base = smem + buf * 512;
        async_ld128(Kb + (size_t)(kt0 + krow) * 64 + kc, base + tid);
        async_ld128(Vb + (size_t)vrow * 2048 + kt0 + vc, base + 256 + tid);
    };
    issue_kv(0, 0);
    int buf = 0;
#else
    uint4 kreg, vreg;
    auto gload_kv = [&](int kt0) {
        kreg = *(const uint4*)(Kb + (size_t)(kt0 + krow) * 64 + kc);
        vreg = *(const uint4*)(Vb + (size_t)vrow * 2048 + kt0 + vc);
    };
    gload_kv(0);
#endif

    v8f oacc[4] = {};
    float m_run = -1e30f, l_run = 0.0f;

    for (int it = 0; it < niter; ++it) {
        const int kt0 = it * 32;
#if HAVE_ASYNC_LDS
        __builtin_amdgcn_s_wait_asynccnt(0);
        __syncthreads();
        const bf16_t* Ks = (const bf16_t*)(smem + buf * 512);
        const bf16_t* Vs = (const bf16_t*)(smem + buf * 512 + 256);
        if (it + 1 < niter) issue_kv(buf ^ 1, kt0 + 32);
        buf ^= 1;
#else
        __syncthreads();
        smem[tid]       = kreg;
        smem[256 + tid] = vreg;
        __syncthreads();
        const bf16_t* Ks = (const bf16_t*)smem;
        const bf16_t* Vs = (const bf16_t*)(smem + 256);
        if (it + 1 < niter) gload_kv(kt0 + 32);
#endif

        // Load ALL K A-fragments first, then run the 4 score WMMAs back-to-back
        Frag ka[2][2];
#pragma unroll
        for (int c2 = 0; c2 < 2; c2++) {
            int ktl = c2 * 16 + l15;
#pragma unroll
            for (int s = 0; s < 2; s++) {
                int dbase = s * 32 + hl * 8;
                ka[c2][s].q[0] = *(const uint4*)(Ks + ktl * 64 + dbase);
                ka[c2][s].q[1] = *(const uint4*)(Ks + ktl * 64 + dbase + 16);
            }
        }
        // Prefetch V^T A-fragments for the PV stage as well (batched ds_loads)
        Frag va[4];
#pragma unroll
        for (int dt = 0; dt < 4; dt++) {
            int drow = dt * 16 + l15;
            int base = hl * 16;
            va[dt].q[0] = *(const uint4*)(Vs + drow * 32 + base);
            va[dt].q[1] = *(const uint4*)(Vs + drow * 32 + base + 8);
        }

        // S^T tiles (kt halves c2=0,1): A = K rows, B = Q^T
        v8f st[2];
#pragma unroll
        for (int c2 = 0; c2 < 2; c2++) {
            st[c2] = (v8f){};
#pragma unroll
            for (int s = 0; s < 2; s++)
                st[c2] = wmma_bf16(ka[c2][s], qf[s], st[c2]);
        }

        // scale 1/sqrt(64), causal mask, per-lane partial row max
        float p0[8], p1[8];
        float pm = -1e30f;
#pragma unroll
        for (int r = 0; r < 8; r++) {
            int kt_a = kt0 + r + hl * 8;
            int kt_b = kt_a + 16;
            float va2 = st[0][r] * 0.125f; if (kt_a > qlane) va2 = -1e30f;
            float vb2 = st[1][r] * 0.125f; if (kt_b > qlane) vb2 = -1e30f;
            p0[r] = va2; p1[r] = vb2;
            pm = fmaxf(pm, fmaxf(va2, vb2));
        }
        pm = fmaxf(pm, __shfl_xor(pm, 16, 32));
        float m_new = fmaxf(m_run, pm);
        float alpha = __expf(m_run - m_new);
        float psum = 0.0f;
#pragma unroll
        for (int r = 0; r < 8; r++) {
            p0[r] = __expf(p0[r] - m_new);
            p1[r] = __expf(p1[r] - m_new);
            psum += p0[r] + p1[r];
        }
        psum += __shfl_xor(psum, 16, 32);
        l_run = l_run * alpha + psum;
        m_run = m_new;
#pragma unroll
        for (int i = 0; i < 4; i++)
#pragma unroll
            for (int r = 0; r < 8; r++) oacc[i][r] *= alpha;

        // Build P^T B-fragment from the two S^T D-tiles (cross-half shuffles)
        Frag pb;
#pragma unroll
        for (int j = 0; j < 8; j++) {
            float x0 = __shfl_xor(p0[j], 16, 32);
            float x1 = __shfl_xor(p1[j], 16, 32);
            float lo = (lane < 16) ? p0[j] : x1;   // element j   : K = base + j
            float hi = (lane < 16) ? x0 : p1[j];   // element j+8 : K = base + 8 + j
            pb.v[j]     = (bf16_t)lo;
            pb.v[j + 8] = (bf16_t)hi;
        }

        // O^T += V^T * P^T (fragments already resident)
#pragma unroll
        for (int dt = 0; dt < 4; dt++)
            oacc[dt] = wmma_bf16(va[dt], pb, oacc[dt]);
    }

    float rl = 1.0f / l_run;
#pragma unroll
    for (int dt = 0; dt < 4; dt++) {
#pragma unroll
        for (int r = 0; r < 8; r++) {
            int d = dt * 16 + r + hl * 8;
            Y[((size_t)(b * 2048) + qlane) * 1024 + h * 64 + d] = (bf16_t)(oacc[dt][r] * rl);
        }
    }
}

// ---------------------------------------------------------------------------
// Host launcher
// ---------------------------------------------------------------------------
extern "C" void kernel_launch(void* const* d_in, const int* in_sizes, int n_in,
                              void* d_out, int out_size, void* d_ws, size_t ws_size,
                              hipStream_t stream) {
    const float* x     = (const float*)d_in[0];  // [2,2048,1024]
    const float* Wqkv  = (const float*)d_in[1];  // [1024,3072]
    const float* Wproj = (const float*)d_in[2];  // [1024,1024]
    float* out = (float*)d_out;                  // [2,2048,1024]

    char* ws = (char*)d_ws;
    size_t off = 0;
    auto carve = [&](size_t elems) -> bf16_t* {
        bf16_t* p = (bf16_t*)(ws + off);
        off += ((elems * sizeof(bf16_t)) + 255) & ~(size_t)255;
        return p;
    };
    const size_t BT = 4096;                      // B*T rows
    bf16_t* xb   = carve(BT * 1024);             // x bf16
    bf16_t* wqT  = carve((size_t)3072 * 1024);   // Wqkv^T [N,K]
    bf16_t* wpT  = carve((size_t)1024 * 1024);   // Wproj^T [N,K]
    bf16_t* qkv  = carve(BT * 3072);             // qkv bf16
    bf16_t* Qr   = carve(BT * 1024);             // Q roped [b,h,t,d]
    bf16_t* Kr   = carve(BT * 1024);             // K roped [b,h,t,d]
    bf16_t* Vt   = carve(BT * 1024);             // V^T    [b,h,d,t]
    bf16_t* Yb   = carve(BT * 1024);             // attention output bf16

    // 1) convert x
    k_f32_to_bf16<<<(int)(BT * 1024 / (256 * 4)), 256, 0, stream>>>(x, xb, (int)(BT * 1024));
    // 2) transpose+convert weights
    k_w_transpose_bf16<<<(3072 * 1024) / 256, 256, 0, stream>>>(Wqkv, wqT, 3072);
    k_w_transpose_bf16<<<(1024 * 1024) / 256, 256, 0, stream>>>(Wproj, wpT, 1024);
    // 3) QKV GEMM: [4096,1024] x [1024,3072] -> bf16
    k_gemm_bf16<true><<<dim3(3072 / 128, 4096 / 128), 256, 0, stream>>>(
        xb, wqT, (void*)qkv, 4096, 3072, 1024);
    // 4) RoPE + scatter to attention layouts
    k_rope_scatter<<<(2 * 2048 * 16 * 32) / 256, 256, 0, stream>>>(qkv, Qr, Kr, Vt);
    // 5) causal flash attention
    k_attn<<<dim3(2048 / 128, 2 * 16), 256, 0, stream>>>(Qr, Kr, Vt, Yb);
    // 6) output projection: [4096,1024] x [1024,1024] -> fp32 d_out
    k_gemm_bf16<false><<<dim3(1024 / 128, 4096 / 128), 256, 0, stream>>>(
        Yb, wpT, (void*)out, 4096, 1024, 1024);
}
